// SeqStructBlock_85272280695390
// MI455X (gfx1250) — compile-verified, matrix operand order checked by reading, not run on hardware
//
#include <hip/hip_runtime.h>
#include <hip/hip_bf16.h>

// ---------------------------------------------------------------------------
// Types / helpers
// ---------------------------------------------------------------------------
typedef __attribute__((ext_vector_type(16))) _Float16 v16h;
typedef __attribute__((ext_vector_type(8)))  _Float16 v8h;
typedef __attribute__((ext_vector_type(8)))  float    v8f;
typedef __attribute__((ext_vector_type(4)))  unsigned int u32x4;

#define CAT16(lo, hi) __builtin_shufflevector((lo), (hi), \
    0,1,2,3,4,5,6,7,8,9,10,11,12,13,14,15)

static __device__ __forceinline__ v8f wmma_f16(v16h a, v16h b, v8f c) {
  // emits v_wmma_f32_16x16x32_f16 (probe-verified builtin, 8 args)
  return __builtin_amdgcn_wmma_f32_16x16x32_f16(false, a, false, b,
                                                (short)0, c, false, false);
}

// Load a 16-half fragment for one lane from a row-major f16 row pointer.
// Lane layout (ISA 7.12.2, 16-bit A 16x32): elems 0..7 = K[k0..k0+7],
// elems 8..15 = K[k0+16..k0+23], where k0 = kb*32 + (lane>=16 ? 8 : 0).
static __device__ __forceinline__ v16h load_frag(const _Float16* rowp, int k0) {
  const v8h* p = (const v8h*)(rowp + k0);
  return CAT16(p[0], p[2]);
}

static __device__ __forceinline__ float sigmoid_f(float x) {
  return 1.0f / (1.0f + __expf(-x));
}
static __device__ __forceinline__ float tanh_f(float x) {
  float xc = fminf(fmaxf(x, -15.f), 15.f);
  float e  = __expf(2.f * xc);
  return (e - 1.f) / (e + 1.f);
}

enum : int { Bz = 8, Lz = 1024, Dz = 256, Hz = 128, Gz = 512,
             Nz = 20000, Ez = 262144, E2z = 524288 };

// ---------------------------------------------------------------------------
// Conversion / init kernels
// ---------------------------------------------------------------------------
__global__ void k_f32_to_f16(const float* __restrict__ s,
                             _Float16* __restrict__ d, int n) {
  int i = blockIdx.x * blockDim.x + threadIdx.x;
  if (i < n) d[i] = (_Float16)s[i];
}

// W (256 x 515) -> f16 (256 x 512), dropping the 3 edge-attr columns
__global__ void k_w515_to_f16(const float* __restrict__ W,
                              _Float16* __restrict__ d) {
  int i = blockIdx.x * blockDim.x + threadIdx.x;
  if (i < 256 * 512) {
    int r = i >> 9, c = i & 511;
    d[i] = (_Float16)W[r * 515 + c];
  }
}

__global__ void k_copy_f32(const float* __restrict__ s,
                           float* __restrict__ d, int n) {
  int i = blockIdx.x * blockDim.x + threadIdx.x;
  if (i < n) d[i] = s[i];
}

// ---------------------------------------------------------------------------
// pre = x @ w_ih^T + b_ih + b_hh   (both directions)
// A: seq_f16 (8192 x 256), B: w_ih_f16[dir] stored [n(512)][k(256)]
// grid (512 Mtiles, 2 dirs), block 256 = 8 waves, 4 N-tiles/wave
// ---------------------------------------------------------------------------
__global__ __launch_bounds__(256) void k_gemm_pre(
    const _Float16* __restrict__ xh,
    const _Float16* __restrict__ wih,
    const float* __restrict__ b_ih_f, const float* __restrict__ b_hh_f,
    const float* __restrict__ b_ih_r, const float* __restrict__ b_hh_r,
    float* __restrict__ pre) {
  const int dir  = blockIdx.y;
  const int mt   = blockIdx.x;
  const int wid  = threadIdx.x >> 5;
  const int lane = threadIdx.x & 31;
  const int m    = lane & 15, hi = lane >> 4;

  const _Float16* w  = wih + (size_t)dir * Gz * Dz;
  const float*    bi = dir ? b_ih_r : b_ih_f;
  const float*    bh = dir ? b_hh_r : b_hh_f;
  const int       row = mt * 16 + m;
  const _Float16* arow = xh + (size_t)row * Dz;

  // hoist A fragments (reused by all 4 N-tiles of this wave)
  v16h afr[8];
#pragma unroll
  for (int kb = 0; kb < 8; ++kb) afr[kb] = load_frag(arow, kb * 32 + hi * 8);

  float* po = pre + (size_t)dir * 8192 * Gz;
#pragma unroll
  for (int ti = 0; ti < 4; ++ti) {
    const int n = (wid * 4 + ti) * 16 + m;
    const _Float16* brow = w + (size_t)n * Dz;
    v8f acc = {};
#pragma unroll
    for (int kb = 0; kb < 8; ++kb)
      acc = wmma_f16(afr[kb], load_frag(brow, kb * 32 + hi * 8), acc);
    const float bias = bi[n] + bh[n];
#pragma unroll
    for (int r = 0; r < 8; ++r) {
      const int rr = mt * 16 + r + hi * 8;      // C layout: M = r + 8*(lane>=16)
      po[(size_t)rr * Gz + n] = acc[r] + bias;
    }
  }
}

// ---------------------------------------------------------------------------
// LSTM recurrence: one workgroup per direction, 32 waves.
// Per step: wave w -> gate columns [w*16, w*16+16), 4 WMMAs over K=128.
// B-fragments (this wave's 16 rows of w_hh) are LOOP-INVARIANT and hoisted
// into registers before the time loop, so the serial critical path contains
// no global loads: ds_load A -> 4x WMMA -> ds_store -> barrier -> VALU cell.
// ---------------------------------------------------------------------------
__global__ __launch_bounds__(1024) void k_lstm(
    const float* __restrict__ pre,      // [2][8192][512]
    const _Float16* __restrict__ whh,   // [2][512][128]  (n-major)
    float* __restrict__ hbuf) {         // [2][8192][128]
  __shared__ __align__(16) _Float16 hsm[16 * Hz];   // h, rows 8..15 stay zero
  __shared__ __align__(16) float    gsm[16 * Gz];   // wmma gate outputs

  const int dir  = blockIdx.x;
  const int tid  = threadIdx.x;
  const int wid  = tid >> 5, lane = tid & 31;
  const int m    = lane & 15, hi = lane >> 4;
  const int b    = tid >> 7, j = tid & 127;
  const int n    = wid * 16 + m;
  const _Float16* brow = whh + ((size_t)dir * Gz + n) * Hz;

  // hoist recurrent-weight fragments into registers (4 x v16h = 32 VGPRs)
  v16h bfr[4];
#pragma unroll
  for (int kb = 0; kb < 4; ++kb) bfr[kb] = load_frag(brow, kb * 32 + hi * 8);

  for (int i = tid; i < 16 * Hz; i += 1024) hsm[i] = (_Float16)0.f;
  __syncthreads();

  float c = 0.f;
  const float* preD = pre + (size_t)dir * 8192 * Gz;
  float* hD = hbuf + (size_t)dir * 8192 * Hz;

  for (int s = 0; s < Lz; ++s) {
    const int t = dir ? (Lz - 1 - s) : s;

    // --- gates_tile = h @ w_hh^T (16 columns of this wave) ---
    v8f acc = {};
#pragma unroll
    for (int kb = 0; kb < 4; ++kb)
      acc = wmma_f16(load_frag(hsm + m * Hz, kb * 32 + hi * 8), bfr[kb], acc);
#pragma unroll
    for (int r = 0; r < 8; ++r)
      gsm[(r + hi * 8) * Gz + n] = acc[r];

    // prefetch next step's pre row for this thread's cell
    if (s + 1 < Lz) {
      const int tn = dir ? (t - 1) : (t + 1);
      __builtin_prefetch(preD + ((size_t)b * Lz + tn) * Gz + j, 0, 1);
    }
    __syncthreads();

    // --- cell update (thread-persistent c) ---
    const float* pr = preD + ((size_t)b * Lz + t) * Gz;
    const float gi = gsm[b * Gz + j]           + pr[j];
    const float gf = gsm[b * Gz + Hz + j]      + pr[Hz + j];
    const float gg = gsm[b * Gz + 2 * Hz + j]  + pr[2 * Hz + j];
    const float go = gsm[b * Gz + 3 * Hz + j]  + pr[3 * Hz + j];
    c = sigmoid_f(gf) * c + sigmoid_f(gi) * tanh_f(gg);
    const float hval = sigmoid_f(go) * tanh_f(c);
    hsm[b * Hz + j] = (_Float16)hval;
    hD[((size_t)b * Lz + t) * Hz + j] = hval;
    __syncthreads();
  }
}

// ---------------------------------------------------------------------------
// Edge gates + segment sums.  32 edges/block; A-panel = [struct[src]|struct[dst]]
// staged in LDS (32 x 512 f16).  Each wave owns 2 N-tiles; per K-step the
// W_in/W_out B-fragments are loaded ONCE and applied to both M-tiles (A from
// LDS), so each pair of global B-loads feeds 4 WMMAs.  128 WMMAs per wave.
// Epilogue: sigmoid(+rank-3 attr term + bias), then global_atomic_add_f32
// into both segment sums (st_acc is L2-resident, 20MB).
// ---------------------------------------------------------------------------
#define EPB 32
__global__ __launch_bounds__(256) void k_edges(
    const int*   __restrict__ ei,      // (2, E2) int32
    const float* __restrict__ eattr,   // (E2, 3)
    const _Float16* __restrict__ sth,  // N x 256 f16
    const _Float16* __restrict__ WinH, // 256 x 512 f16
    const _Float16* __restrict__ WoutH,// 256 x 512 f16
    const float* __restrict__ Win,     // 256 x 515 (last 3 cols)
    const float* __restrict__ bin,
    const float* __restrict__ Wout,
    const float* __restrict__ bout,
    float* __restrict__ st) {
  __shared__ __align__(16) _Float16 feat[EPB][512];
  __shared__ int   sSrc[EPB], sDst[EPB];
  __shared__ float aIn[EPB][4], aOut[EPB][4];

  const int tid = threadIdx.x;
  const int e0  = blockIdx.x * EPB;

  { // stage gathered struct rows: 8 threads/edge, 64 f16 each
    const int el = tid >> 3, q = tid & 7;
    const int e  = e0 + el;
    const int srcI = ei[2 * e];
    const int dstI = ei[E2z + 2 * e];
    const _Float16* rp = sth + (size_t)(q < 4 ? srcI : dstI) * Dz + (q & 3) * 64;
    u32x4*       dp = (u32x4*)&feat[el][q * 64];
    const u32x4* sp = (const u32x4*)rp;
#pragma unroll
    for (int i = 0; i < 8; ++i) dp[i] = sp[i];
  }
  if (tid < EPB) {
    const int e = e0 + tid;
    sSrc[tid] = ei[2 * e];
    sDst[tid] = ei[E2z + 2 * e];
    aIn[tid][0]  = eattr[(2 * e + 1) * 3 + 0];
    aIn[tid][1]  = eattr[(2 * e + 1) * 3 + 1];
    aIn[tid][2]  = eattr[(2 * e + 1) * 3 + 2];
    aOut[tid][0] = eattr[(2 * e) * 3 + 0];
    aOut[tid][1] = eattr[(2 * e) * 3 + 1];
    aOut[tid][2] = eattr[(2 * e) * 3 + 2];
  }
  __syncthreads();

  const int wid = tid >> 5, lane = tid & 31;
  const int m = lane & 15, hi = lane >> 4;

#pragma unroll
  for (int ni = 0; ni < 2; ++ni) {
    const int nt = wid + ni * 8;               // 0..15
    const int n  = nt * 16 + m;
    const _Float16* bi_row = WinH  + (size_t)n * Gz;
    const _Float16* bo_row = WoutH + (size_t)n * Gz;
    const _Float16* arow0  = &feat[m][0];      // M-tile 0 (edges 0..15)
    const _Float16* arow1  = &feat[16 + m][0]; // M-tile 1 (edges 16..31)

    v8f aci0 = {}, aco0 = {}, aci1 = {}, aco1 = {};
#pragma unroll 4
    for (int kb = 0; kb < 16; ++kb) {
      const int k0 = kb * 32 + hi * 8;
      const v16h bIn  = load_frag(bi_row, k0);   // loaded once,
      const v16h bOut = load_frag(bo_row, k0);   // reused by both M-tiles
      const v16h a0 = load_frag(arow0, k0);
      aci0 = wmma_f16(a0, bIn,  aci0);
      aco0 = wmma_f16(a0, bOut, aco0);
      const v16h a1 = load_frag(arow1, k0);
      aci1 = wmma_f16(a1, bIn,  aci1);
      aco1 = wmma_f16(a1, bOut, aco1);
    }

    // epilogue: rank-3 attr columns + bias, sigmoid, dual segment-sum atomics
    const float wi0 = Win[n * 515 + 512],  wi1 = Win[n * 515 + 513];
    const float wi2 = Win[n * 515 + 514],  bI  = bin[n];
    const float wo0 = Wout[n * 515 + 512], wo1 = Wout[n * 515 + 513];
    const float wo2 = Wout[n * 515 + 514], bO  = bout[n];
#pragma unroll
    for (int mt = 0; mt < 2; ++mt) {
      const v8f aci = mt ? aci1 : aci0;
      const v8f aco = mt ? aco1 : aco0;
#pragma unroll
      for (int r = 0; r < 8; ++r) {
        const int el = mt * 16 + r + hi * 8;
        float gi = aci[r] + aIn[el][0]*wi0 + aIn[el][1]*wi1 + aIn[el][2]*wi2 + bI;
        float go = aco[r] + aOut[el][0]*wo0 + aOut[el][1]*wo1 + aOut[el][2]*wo2 + bO;
        gi = sigmoid_f(gi);
        go = sigmoid_f(go);
        const float snd = (float)feat[el][256 + n];   // sn_dst[el][n]
        const float sns = (float)feat[el][n];         // sn_src[el][n]
        atomicAdd(&st[(size_t)sSrc[el] * Dz + n], gi * snd);
        atomicAdd(&st[(size_t)sDst[el] * Dz + n], go * sns);
      }
    }
  }
}

// ---------------------------------------------------------------------------
// LayerNorm kernels: one wave per 256-wide row, shfl_xor reductions (wave32)
// ---------------------------------------------------------------------------
__global__ __launch_bounds__(256) void k_ln_seq(
    const float* __restrict__ seq, const float* __restrict__ hbuf,
    const float* __restrict__ g, const float* __restrict__ bb,
    float* __restrict__ out) {
  const int wid = threadIdx.x >> 5, lane = threadIdx.x & 31;
  const int row = blockIdx.x * 8 + wid;                 // 0..8191
  const float* hf = hbuf + (size_t)row * Hz;
  const float* hb = hbuf + (size_t)8192 * Hz + (size_t)row * Hz;
  float v[8], sum = 0.f, sq = 0.f;
#pragma unroll
  for (int i = 0; i < 8; ++i) {
    const int d = i * 32 + lane;
    float x = seq[(size_t)row * Dz + d] + (d < Hz ? hf[d] : hb[d - Hz]);
    v[i] = x; sum += x; sq += x * x;
  }
#pragma unroll
  for (int o = 16; o > 0; o >>= 1) {
    sum += __shfl_xor(sum, o);
    sq  += __shfl_xor(sq, o);
  }
  const float mean = sum * (1.f / Dz);
  const float rstd = rsqrtf(sq * (1.f / Dz) - mean * mean + 1e-5f);
#pragma unroll
  for (int i = 0; i < 8; ++i) {
    const int d = i * 32 + lane;
    out[(size_t)row * Dz + d] = (v[i] - mean) * rstd * g[d] + bb[d];
  }
}

__global__ __launch_bounds__(256) void k_ln_struct(
    const float* __restrict__ stacc,
    const float* __restrict__ g, const float* __restrict__ bb,
    float* __restrict__ out) {
  const int wid = threadIdx.x >> 5, lane = threadIdx.x & 31;
  const int row = blockIdx.x * 8 + wid;                 // 0..19999
  float v[8], sum = 0.f, sq = 0.f;
#pragma unroll
  for (int i = 0; i < 8; ++i) {
    const int d = i * 32 + lane;
    float x = stacc[(size_t)row * Dz + d];
    v[i] = x; sum += x; sq += x * x;
  }
#pragma unroll
  for (int o = 16; o > 0; o >>= 1) {
    sum += __shfl_xor(sum, o);
    sq  += __shfl_xor(sq, o);
  }
  const float mean = sum * (1.f / Dz);
  const float rstd = rsqrtf(sq * (1.f / Dz) - mean * mean + 1e-5f);
#pragma unroll
  for (int i = 0; i < 8; ++i) {
    const int d = i * 32 + lane;
    out[(size_t)row * Dz + d] = (v[i] - mean) * rstd * g[d] + bb[d];
  }
}

// ---------------------------------------------------------------------------
// Host launcher
// ---------------------------------------------------------------------------
extern "C" void kernel_launch(void* const* d_in, const int* in_sizes, int n_in,
                              void* d_out, int out_size, void* d_ws, size_t ws_size,
                              hipStream_t stream) {
  (void)in_sizes; (void)n_in; (void)out_size; (void)ws_size;
  const float* seq    = (const float*)d_in[0];
  const float* stru   = (const float*)d_in[1];
  const int*   ei     = (const int*)d_in[2];     // int32 (JAX default x64 off)
  const float* eattr  = (const float*)d_in[3];
  // d_in[4] middleSelect: unused by the reference
  const float* w_ih_f = (const float*)d_in[5];
  const float* w_hh_f = (const float*)d_in[6];
  const float* b_ih_f = (const float*)d_in[7];
  const float* b_hh_f = (const float*)d_in[8];
  const float* w_ih_r = (const float*)d_in[9];
  const float* w_hh_r = (const float*)d_in[10];
  const float* b_ih_r = (const float*)d_in[11];
  const float* b_hh_r = (const float*)d_in[12];
  const float* g_seq  = (const float*)d_in[13];
  const float* b_seq  = (const float*)d_in[14];
  const float* W_in   = (const float*)d_in[15];
  const float* b_in   = (const float*)d_in[16];
  const float* W_out  = (const float*)d_in[17];
  const float* b_out  = (const float*)d_in[18];
  const float* g_st   = (const float*)d_in[19];
  const float* b_st   = (const float*)d_in[20];

  // workspace layout
  char* w = (char*)d_ws;
  size_t off = 0;
  auto take = [&](size_t bytes) { char* p = w + off; off = (off + bytes + 255) & ~(size_t)255; return p; };
  _Float16* seqH  = (_Float16*)take((size_t)8192 * 256 * 2);
  _Float16* wihH  = (_Float16*)take((size_t)2 * 512 * 256 * 2);
  _Float16* whhH  = (_Float16*)take((size_t)2 * 512 * 128 * 2);
  _Float16* stH   = (_Float16*)take((size_t)20000 * 256 * 2);
  _Float16* winH  = (_Float16*)take((size_t)256 * 512 * 2);
  _Float16* woutH = (_Float16*)take((size_t)256 * 512 * 2);
  float*    pre   = (float*)take((size_t)2 * 8192 * 512 * 4);
  float*    hbuf  = (float*)take((size_t)2 * 8192 * 128 * 4);
  float*    stAcc = (float*)take((size_t)20000 * 256 * 4);

  float* outSeq = (float*)d_out;                       // 8*1024*256
  float* outSt  = outSeq + (size_t)8192 * 256;         // 20000*256

  auto blocks = [](int n) { return (n + 255) / 256; };

  // 1) precision conversion + st accumulator init
  k_f32_to_f16<<<blocks(8192 * 256), 256, 0, stream>>>(seq, seqH, 8192 * 256);
  k_f32_to_f16<<<blocks(512 * 256), 256, 0, stream>>>(w_ih_f, wihH, 512 * 256);
  k_f32_to_f16<<<blocks(512 * 256), 256, 0, stream>>>(w_ih_r, wihH + (size_t)512 * 256, 512 * 256);
  k_f32_to_f16<<<blocks(512 * 128), 256, 0, stream>>>(w_hh_f, whhH, 512 * 128);
  k_f32_to_f16<<<blocks(512 * 128), 256, 0, stream>>>(w_hh_r, whhH + (size_t)512 * 128, 512 * 128);
  k_f32_to_f16<<<blocks(20000 * 256), 256, 0, stream>>>(stru, stH, 20000 * 256);
  k_w515_to_f16<<<blocks(256 * 512), 256, 0, stream>>>(W_in, winH);
  k_w515_to_f16<<<blocks(256 * 512), 256, 0, stream>>>(W_out, woutH);
  k_copy_f32<<<blocks(20000 * 256), 256, 0, stream>>>(stru, stAcc, 20000 * 256);

  // 2) LSTM input GEMM (WMMA)
  k_gemm_pre<<<dim3(512, 2), 256, 0, stream>>>(seqH, wihH,
      b_ih_f, b_hh_f, b_ih_r, b_hh_r, pre);

  // 3) LSTM recurrence (WMMA per step, weights resident in VGPRs)
  k_lstm<<<2, 1024, 0, stream>>>(pre, whhH, hbuf);

  // 4) edge gate GEMMs + segment-sum atomics (WMMA + global_atomic_add_f32)
  k_edges<<<Ez / EPB, 256, 0, stream>>>(ei, eattr, stH, winH, woutH,
      W_in, b_in, W_out, b_out, stAcc);

  // 5) LayerNorms -> outputs
  k_ln_seq<<<8192 / 8, 256, 0, stream>>>(seq, hbuf, g_seq, b_seq, outSeq);
  k_ln_struct<<<20000 / 8, 256, 0, stream>>>(stAcc, g_st, b_st, outSt);
}
